// MultiHeadAttention2D_65601330479141
// MI455X (gfx1250) — matrix operand order repaired and checked by reading, not compiled, not run on hardware
//
#include <hip/hip_runtime.h>

// ============================================================================
// MultiHeadAttention2D (CompressAI-style) for MI455X / gfx1250 (CDNA5).
//
// Compute-bound (~175 GFLOP): all convs are implicit GEMMs through
// v_wmma_f32_16x16x32_f16 (wave32, f32 accum). Activations f16 [img][pix][chan].
// Weights are re-packed once per launch into f16 WMMA A-fragment order, so an
// A fragment is ONE aligned 32B load. Activation slabs (128KB/image) are staged
// global->LDS with the Tensor Data Mover (tensor_load_to_lds + s_wait_tensorcnt).
// Residual units are fused one-image-per-workgroup in 192KB of the 320KB WGP LDS.
//
// Input flattening assumption (setup_inputs() dict insertion order, DFS):
//  0 query 1 k 2 v | 3..8 q_w,q_b,k_w,k_b,v_w,v_b
//  9..44 att: ru_a[0..2], ru_b[0..2] each (w1,b1,w2,b2,w3,b3)
//  45 b_out_w 46 b_out_b | 47..52 res (w1,b1,w2,b2,skip_w,skip_b) | 53 out_w 54 out_b
//
// Workspace (~128MB): see kernel_launch for the layout.
// ============================================================================

typedef __attribute__((ext_vector_type(16))) _Float16 v16h;
typedef __attribute__((ext_vector_type(8)))  float    v8f;
typedef unsigned int u32x4 __attribute__((ext_vector_type(4)));
typedef int          i32x8 __attribute__((ext_vector_type(8)));
typedef int          i32x4 __attribute__((ext_vector_type(4)));

#define IMG_H 16
#define HW    256   // pixels per image

// ---------------------------------------------------------------------------
// TDM: DMA `bytes` (multiple of 8) from global `gsrc` into LDS offset `ldsOff`.
// 1-D tensor of 8-byte elements, tile == tensor (n8 <= 65535 fits tile_dim0).
// D# per CDNA5 ISA; groups 2/3 zero (<=2D tensor).
// This toolchain exposes the 6-arg builtin: (g0, g1, g2, g3, g_extra, cpol).
// Wave-level op (EXEC ignored): issue from ONE wave, s_wait_tensorcnt, barrier.
// ---------------------------------------------------------------------------
__device__ __forceinline__ void tdm_load_to_lds(const void* gsrc,
                                                unsigned ldsOff, unsigned bytes) {
  const unsigned n8 = bytes >> 3;
  const unsigned long long ga = (unsigned long long)gsrc;
  u32x4 g0 = {};
  g0[0] = 1u;                                            // count=1 (valid user D#)
  g0[1] = ldsOff;                                        // lds_addr (bytes)
  g0[2] = (unsigned)(ga & 0xffffffffu);                  // global_addr[31:0]
  g0[3] = (unsigned)((ga >> 32) & 0x01ffffffu)           // global_addr[56:32]
          | (2u << 30);                                  // type = 2 ("image")
  i32x8 g1 = {};
  g1[0] = (3 << 16);                                     // data_size=8B, mask=0
  g1[1] = (int)((n8 & 0xffffu) << 16);                   // tensor_dim0[15:0]
  g1[2] = (int)((n8 >> 16) & 0xffffu) | (1 << 16);       // dim0[31:16], dim1=1
  g1[3] = (int)((n8 & 0xffffu) << 16);                   // tile_dim0
  g1[4] = 0;                                             // tile_dim1/2 unused
  g1[5] = (int)n8;                                       // tensor_dim0_stride lo
  g1[6] = 0; g1[7] = 0;
  i32x4 z4 = {};
  i32x8 z8 = {};
  __builtin_amdgcn_tensor_load_to_lds(g0, g1, z4, z4, z8, 0);
}

// ---------------------------------------------------------------------------
// Pack fp32 OIHW weights -> f16 in WMMA A-fragment lane order.
// Fragment (mt, tap, kc): 32 lanes x 16 halfs, laid out so the consuming lane
// does one contiguous 32B load. A-matrix layout used here (16-bit, 16x32, contiguous k):
//   m = lane&15; lanes 0-15: K0..15, lanes 16-31: K16..31 (matches B)
// Rows m >= Cout are zero-filled (Cout=8 residual-path convs).
// ---------------------------------------------------------------------------
__global__ __launch_bounds__(256) void pack_weights(
    const float* __restrict__ W, _Float16* __restrict__ out,
    int Cout, int Cin, int KK) {
  const int MT = (Cout + 15) >> 4, KT = Cin >> 5;
  const int total = MT * KK * KT * 32;
  for (int i = blockIdx.x * blockDim.x + threadIdx.x; i < total;
       i += gridDim.x * blockDim.x) {
    const int lane = i & 31;
    int rest = i >> 5;
    const int kc = rest % KT; rest /= KT;
    const int tap = rest % KK; rest /= KK;
    const int mt = rest;
    const int m = mt * 16 + (lane & 15);
    const int khalf = (lane & 16) ? 16 : 0;
    v16h a = {};
    if (m < Cout) {
#pragma unroll
      for (int v = 0; v < 8; ++v) {
        const int kb = 2 * v + khalf;
        const int k = kc * 32 + kb;
        a[2 * v]     = (_Float16)W[((size_t)m * Cin + k)     * KK + tap];
        a[2 * v + 1] = (_Float16)W[((size_t)m * Cin + k + 1) * KK + tap];
      }
    }
    *(v16h*)(out + (size_t)i * 16) = a;
  }
}

// A fragment = one 32B load from packed weights.
__device__ __forceinline__ v16h load_a_packed(const _Float16* __restrict__ P,
                                              int mt, int KK, int KT, int tap,
                                              int kc, int lane) {
  return *(const v16h*)(P + ((((size_t)mt * KK + tap) * KT + kc) * 32 + lane) * 16);
}

// ---------------------------------------------------------------------------
// B-fragment: 32x16 f16 from LDS activations act[pix][Cs] (chan contiguous).
// Layout used here (contiguous k, same as A): n = lane&15; lanes 0-15 hold K0..15, lanes 16-31 hold K16..31
// -> one contiguous 32B-aligned read (ds_load_b128 pair). 'dx' = 3x3 x-shift.
// ---------------------------------------------------------------------------
__device__ __forceinline__ v16h load_b_frag(const _Float16* __restrict__ act,
                                            int Cs, int yy, int k0, int lane,
                                            int dx) {
  const int n = lane & 15;
  const int x = n + dx;
  const int khalf = (lane & 16) ? 16 : 0;
  v16h b = {};
  if ((unsigned)x < 16u) {
    const _Float16* p = act + (size_t)(yy * IMG_H + x) * Cs + k0 + khalf;
    b = *(const v16h*)p;
  }
  return b;
}

// C layout: VGPR r -> M = m0 + r + 8*(lane>=16), N = lane&15.
__device__ __forceinline__ void conv_epilogue(
    const v8f& c, int img, int m0, int y, int lane, int Cout,
    const float* __restrict__ bias, const _Float16* __restrict__ skip,
    void* __restrict__ out, int actMode, int outMode) {
  const int hi = (lane >> 4) & 1;
  const int n  = lane & 15;
  const int pix = y * IMG_H + n;
#pragma unroll
  for (int r = 0; r < 8; ++r) {
    const int m = m0 + r + hi * 8;
    if (m >= Cout) continue;
    float v = c[r] + bias[m];
    if (skip) v += (float)skip[(size_t)img * HW * Cout + (size_t)pix * Cout + m];
    if (actMode == 1)      v = fmaxf(v, 0.f);
    else if (actMode == 2) v = v > 0.f ? v : 0.01f * v;
    if (outMode == 0)
      ((_Float16*)out)[(size_t)img * HW * Cout + (size_t)pix * Cout + m] = (_Float16)v;
    else if (outMode == 1)
      ((float*)out)[(size_t)img * HW * Cout + (size_t)pix * Cout + m] = v;
    else
      ((float*)out)[((size_t)img * Cout + m) * HW + pix] = v;          // NCHW
  }
}

// ---------------------------------------------------------------------------
// Generic WMMA conv (1x1 or 3x3 'same'), one image per block, 8 wave32 waves.
// in: f16 [img][pix][Cin] (Cin==256 at every call site). Wp: packed f16.
// actMode: 0 none, 1 relu, 2 leaky(0.01)
// outMode: 0 f16 [img][pix][Cout], 1 f32 [img][pix][Cout], 2 f32 NCHW
// ---------------------------------------------------------------------------
__global__ __launch_bounds__(256) void conv_wmma(
    const _Float16* __restrict__ in, const _Float16* __restrict__ Wp,
    const float* __restrict__ bias, const _Float16* __restrict__ skip,
    void* __restrict__ out, int Cin, int Cout, int k3, int actMode, int outMode) {
  extern __shared__ _Float16 smem[];             // [HW][Cin] staged activations
  const int img  = blockIdx.x;
  const int tid  = threadIdx.x;
  const int lane = tid & 31;
  const int wave = tid >> 5;
  const int nw   = (int)(blockDim.x >> 5);
  const int KK   = k3 ? 9 : 1;
  const int MT   = (Cout + 15) >> 4;
  const int KT   = Cin >> 5;

  // Stage this image's activations into LDS via the Tensor Data Mover.
  if (wave == 0) {
    tdm_load_to_lds(in + (size_t)img * HW * Cin,
                    (unsigned)__builtin_amdgcn_groupstaticsize(),
                    (unsigned)(HW * Cin * sizeof(_Float16)));
    __builtin_amdgcn_s_wait_tensorcnt(0);
  } else {
    // Warm packed weights into caches while the DMA runs.
    const size_t wBytes = (size_t)MT * KK * KT * 512 * sizeof(_Float16);
    for (size_t i = (size_t)tid * 256; i < wBytes; i += (size_t)blockDim.x * 256)
      __builtin_prefetch((const char*)Wp + i, 0, 1);
  }
  __syncthreads();

  if (!k3 && Cin == 256 && MT >= nw) {
    // 1x1, Cout>=128: hoist the 8 A fragments per M-tile, reuse over 16 rows.
    for (int mt = wave; mt < MT; mt += nw) {
      v16h areg[8];
#pragma unroll
      for (int kc = 0; kc < 8; ++kc)
        areg[kc] = load_a_packed(Wp, mt, 1, 8, 0, kc, lane);
      for (int y = 0; y < 16; ++y) {
        v8f c = {};
#pragma unroll
        for (int kc = 0; kc < 8; ++kc) {
          v16h b = load_b_frag(smem, 256, y, kc * 32, lane, 0);
          c = __builtin_amdgcn_wmma_f32_16x16x32_f16(false, areg[kc], false, b,
                                                     (short)0, c, false, false);
        }
        conv_epilogue(c, img, mt << 4, y, lane, Cout, bias, skip, out,
                      actMode, outMode);
      }
    }
  } else {
    // Generic path (3x3 taps / small Cout).
    const int totalTiles = MT * IMG_H;
    for (int t = wave; t < totalTiles; t += nw) {
      const int mt = t >> 4;
      const int y  = t & 15;
      v8f c = {};
      for (int tap = 0; tap < KK; ++tap) {
        const int dy = k3 ? (tap / 3 - 1) : 0;
        const int dx = k3 ? (tap % 3 - 1) : 0;
        const int yy = y + dy;
        if ((unsigned)yy >= 16u) continue;       // whole-row tap is zero
        for (int kc = 0; kc < KT; ++kc) {
          v16h a = load_a_packed(Wp, mt, KK, KT, tap, kc, lane);
          v16h b = load_b_frag(smem, Cin, yy, kc * 32, lane, dx);
          c = __builtin_amdgcn_wmma_f32_16x16x32_f16(false, a, false, b,
                                                     (short)0, c, false, false);
        }
      }
      conv_epilogue(c, img, mt << 4, y, lane, Cout, bias, skip, out,
                    actMode, outMode);
    }
  }
}

// ---------------------------------------------------------------------------
// Fused CompressAI ResidualUnit, one image per workgroup, mids in LDS:
//   h = relu(1x1 256->128); h = relu(3x3 128->128); h = 1x1 128->256;
//   out = relu(x + h)
// LDS: Xs [HW][256] (128KB, phase-1 only) | M1 [HW][128] (64KB)
//      M2 overlays Xs after the phase-1 barrier -> 192KB total (<=320KB/WGP).
// Safe for in == out (skip re-read from global by the writing thread).
// ---------------------------------------------------------------------------
__global__ __launch_bounds__(256) void residual_unit_wmma(
    const _Float16* __restrict__ in, _Float16* __restrict__ out,
    const _Float16* __restrict__ w1p, const float* __restrict__ b1,
    const _Float16* __restrict__ w2p, const float* __restrict__ b2,
    const _Float16* __restrict__ w3p, const float* __restrict__ b3) {
  extern __shared__ _Float16 lds[];
  _Float16* Xs = lds;            // 65536 halfs
  _Float16* M1 = lds + 65536;    // 32768 halfs
  _Float16* M2 = lds;            // 32768 halfs (overlays Xs after barrier)
  const int img  = blockIdx.x;
  const int tid  = threadIdx.x;
  const int lane = tid & 31;
  const int wave = tid >> 5;
  const int nw   = (int)(blockDim.x >> 5);
  const _Float16* gin = in + (size_t)img * HW * 256;

  if (wave == 0) {               // TDM-stage X (131072 bytes)
    tdm_load_to_lds(gin, (unsigned)__builtin_amdgcn_groupstaticsize(), 131072u);
    __builtin_amdgcn_s_wait_tensorcnt(0);
  }
  __syncthreads();

  const int hi = (lane >> 4) & 1;
  const int n  = lane & 15;

  // ---- conv1: 1x1 256->128, relu -> M1 (MT=8 == nw: one M-tile per wave) ----
  {
    const int mt = wave;
    v16h areg[8];
#pragma unroll
    for (int kc = 0; kc < 8; ++kc)
      areg[kc] = load_a_packed(w1p, mt, 1, 8, 0, kc, lane);
    for (int y = 0; y < 16; ++y) {
      v8f c = {};
#pragma unroll
      for (int kc = 0; kc < 8; ++kc) {
        v16h b = load_b_frag(Xs, 256, y, kc * 32, lane, 0);
        c = __builtin_amdgcn_wmma_f32_16x16x32_f16(false, areg[kc], false, b,
                                                   (short)0, c, false, false);
      }
      const int pix = y * IMG_H + n;
#pragma unroll
      for (int r = 0; r < 8; ++r) {
        const int m = (mt << 4) + r + hi * 8;
        M1[(size_t)pix * 128 + m] = (_Float16)fmaxf(c[r] + b1[m], 0.f);
      }
    }
  }
  __syncthreads();   // all Xs reads done; M2 may overlay Xs

  // ---- conv2: 3x3 128->128, relu -> M2 ----
  for (int t = wave; t < 128; t += nw) {
    const int mt = t >> 4, y = t & 15;
    v8f c = {};
    for (int tap = 0; tap < 9; ++tap) {
      const int dy = tap / 3 - 1, dx = tap % 3 - 1;
      const int yy = y + dy;
      if ((unsigned)yy >= 16u) continue;
#pragma unroll
      for (int kc = 0; kc < 4; ++kc) {
        v16h a = load_a_packed(w2p, mt, 9, 4, tap, kc, lane);
        v16h b = load_b_frag(M1, 128, yy, kc * 32, lane, dx);
        c = __builtin_amdgcn_wmma_f32_16x16x32_f16(false, a, false, b,
                                                   (short)0, c, false, false);
      }
    }
    const int pix = y * IMG_H + n;
#pragma unroll
    for (int r = 0; r < 8; ++r) {
      const int m = (mt << 4) + r + hi * 8;
      M2[(size_t)pix * 128 + m] = (_Float16)fmaxf(c[r] + b2[m], 0.f);
    }
  }
  __syncthreads();

  // ---- conv3: 1x1 128->256, +x skip, relu -> global out (MT=16) ----
  _Float16* gout = out + (size_t)img * HW * 256;
  for (int mt = wave; mt < 16; mt += nw) {
    v16h areg[4];
#pragma unroll
    for (int kc = 0; kc < 4; ++kc)
      areg[kc] = load_a_packed(w3p, mt, 1, 4, 0, kc, lane);
    for (int y = 0; y < 16; ++y) {
      v8f c = {};
#pragma unroll
      for (int kc = 0; kc < 4; ++kc) {
        v16h b = load_b_frag(M2, 128, y, kc * 32, lane, 0);
        c = __builtin_amdgcn_wmma_f32_16x16x32_f16(false, areg[kc], false, b,
                                                   (short)0, c, false, false);
      }
      const int pix = y * IMG_H + n;
#pragma unroll
      for (int r = 0; r < 8; ++r) {
        const int m = (mt << 4) + r + hi * 8;
        float v = c[r] + b3[m] + (float)gin[(size_t)pix * 256 + m];
        gout[(size_t)pix * 256 + m] = (_Float16)fmaxf(v, 0.f);
      }
    }
  }
}

// fp32 NCHW [img][C][256] -> f16 [img][pix][C]
__global__ __launch_bounds__(256) void pack_nchw_to_phwc(
    const float* __restrict__ in, _Float16* __restrict__ out, int C) {
  const int img = blockIdx.x, pix = threadIdx.x;
  const float* src = in + (size_t)img * C * HW;
  _Float16* dst = out + (size_t)img * HW * C + (size_t)pix * C;
  for (int c = 0; c < C; ++c) dst[c] = (_Float16)src[(size_t)c * HW + pix];
}

// att_in[(b,q,k)] = qact[(b,q)] + kact[(b,k)]   (256 images out)
__global__ __launch_bounds__(256) void pairsum(
    const _Float16* __restrict__ q, const _Float16* __restrict__ k,
    _Float16* __restrict__ out) {
  const int img = blockIdx.x;
  const int b = img >> 6, qi = (img >> 3) & 7, ki = img & 7;
  const _Float16* qs = q + (size_t)(b * 8 + qi) * 65536;
  const _Float16* ks = k + (size_t)(b * 8 + ki) * 65536;
  _Float16* o = out + (size_t)img * 65536;
  for (int i = threadIdx.x; i < 65536; i += (int)blockDim.x)
    o[i] = (_Float16)((float)qs[i] + (float)ks[i]);
}

// b := x + a * sigmoid(b)    (in-place on the gate buffer)
__global__ __launch_bounds__(256) void combine_gate(
    const _Float16* __restrict__ x, const _Float16* __restrict__ a,
    _Float16* __restrict__ b) {
  const size_t base = (size_t)blockIdx.x * 65536;
  for (int i = threadIdx.x; i < 65536; i += (int)blockDim.x) {
    const float g = 1.f / (1.f + expf(-(float)b[base + i]));
    b[base + i] = (_Float16)((float)x[base + i] + (float)a[base + i] * g);
  }
}

// tiny conv3x3 8->8 + leaky, fp32 [img][pix][8] -> same layout
__global__ __launch_bounds__(256) void conv3x3_c8_scalar(
    const float* __restrict__ in, const float* __restrict__ W,
    const float* __restrict__ bias, float* __restrict__ out) {
  const int img = blockIdx.x, pix = threadIdx.x;
  const int y = pix >> 4, x = pix & 15;
  const float* src = in + (size_t)img * HW * 8;
  float o[8];
#pragma unroll
  for (int co = 0; co < 8; ++co) o[co] = bias[co];
  for (int dy = -1; dy <= 1; ++dy) {
    const int yy = y + dy;
    if ((unsigned)yy >= 16u) continue;
    for (int dx = -1; dx <= 1; ++dx) {
      const int xx = x + dx;
      if ((unsigned)xx >= 16u) continue;
      const float* s = src + (size_t)(yy * IMG_H + xx) * 8;
#pragma unroll
      for (int co = 0; co < 8; ++co)
#pragma unroll
        for (int ci = 0; ci < 8; ++ci)
          o[co] += W[((co * 8 + ci) * 3 + (dy + 1)) * 3 + (dx + 1)] * s[ci];
    }
  }
  float* dst = out + (size_t)img * HW * 8 + (size_t)pix * 8;
#pragma unroll
  for (int co = 0; co < 8; ++co)
    dst[co] = o[co] > 0.f ? o[co] : 0.01f * o[co];
}

// s = (r2 + skip) / sqrt(32); softmax over k (KL=8).  One block per (b,q).
__global__ __launch_bounds__(256) void score_softmax(
    const float* __restrict__ r2, const float* __restrict__ sk,
    float* __restrict__ p) {
  const int bq = blockIdx.x, pix = threadIdx.x;
  const float scale = 0.1767766952966369f;  // 1/sqrt(32)
  for (int h = 0; h < 8; ++h) {
    float l[8], mx = -3.4e38f;
#pragma unroll
    for (int k = 0; k < 8; ++k) {
      const size_t idx = ((size_t)(bq * 8 + k) * HW + pix) * 8 + h;
      l[k] = (r2[idx] + sk[idx]) * scale;
      mx = fmaxf(mx, l[k]);
    }
    float sum = 0.f;
#pragma unroll
    for (int k = 0; k < 8; ++k) { l[k] = expf(l[k] - mx); sum += l[k]; }
    const float inv = 1.f / sum;
#pragma unroll
    for (int k = 0; k < 8; ++k) {
      const size_t idx = ((size_t)(bq * 8 + k) * HW + pix) * 8 + h;
      p[idx] = l[k] * inv;
    }
  }
}

// out[(b,q)][pix][c] = sum_k probs[(b,q,k)][pix][c>>5] * vact[(b,k)][pix][c]
__global__ __launch_bounds__(256) void attn_apply(
    const float* __restrict__ p, const _Float16* __restrict__ vact,
    _Float16* __restrict__ out) {
  const int bq = blockIdx.x, pix = threadIdx.x;
  const int b = bq >> 3;
  _Float16* dst = out + (size_t)bq * 65536 + (size_t)pix * 256;
  for (int h = 0; h < 8; ++h) {
    float pk[8];
#pragma unroll
    for (int k = 0; k < 8; ++k)
      pk[k] = p[((size_t)(bq * 8 + k) * HW + pix) * 8 + h];
    for (int d = 0; d < 32; ++d) {
      const int c = h * 32 + d;
      float acc = 0.f;
#pragma unroll
      for (int k = 0; k < 8; ++k)
        acc += pk[k] * (float)vact[(size_t)(b * 8 + k) * 65536 + (size_t)pix * 256 + c];
      dst[c] = (_Float16)acc;
    }
  }
}

// ===========================================================================
extern "C" void kernel_launch(void* const* d_in, const int* in_sizes, int n_in,
                              void* d_out, int out_size, void* d_ws, size_t ws_size,
                              hipStream_t stream) {
  (void)in_sizes; (void)n_in; (void)out_size; (void)ws_size;
  const float* query = (const float*)d_in[0];
  const float* kin   = (const float*)d_in[1];
  const float* vin   = (const float*)d_in[2];
  const float* q_w = (const float*)d_in[3];
  const float* q_b = (const float*)d_in[4];
  const float* k_w = (const float*)d_in[5];
  const float* k_b = (const float*)d_in[6];
  const float* v_w = (const float*)d_in[7];
  const float* v_b = (const float*)d_in[8];
  const float* ru[6][6];                 // ru_a[0..2] then ru_b[0..2]
  int idx = 9;
  for (int u = 0; u < 6; ++u)
    for (int pp = 0; pp < 6; ++pp) ru[u][pp] = (const float*)d_in[idx++];
  const float* b_out_w = (const float*)d_in[idx++];
  const float* b_out_b = (const float*)d_in[idx++];
  const float* r_w1  = (const float*)d_in[idx++];
  const float* r_b1  = (const float*)d_in[idx++];
  const float* r_w2  = (const float*)d_in[idx++];
  const float* r_b2  = (const float*)d_in[idx++];
  const float* r_skw = (const float*)d_in[idx++];
  const float* r_skb = (const float*)d_in[idx++];
  const float* out_w = (const float*)d_in[idx++];
  const float* out_b = (const float*)d_in[idx++];

  char* ws = (char*)d_ws;
  const size_t MB = 1024ull * 1024ull;
  _Float16* PK   = (_Float16*)(ws + 0 * MB);     // packed input (reused q/k/v)
  _Float16* QA   = (_Float16*)(ws + 4 * MB);
  _Float16* KA   = (_Float16*)(ws + 8 * MB);
  _Float16* VA   = (_Float16*)(ws + 12 * MB);
  _Float16* X0   = (_Float16*)(ws + 16 * MB);    // att_in (256 images)
  _Float16* Abuf = (_Float16*)(ws + 48 * MB);
  _Float16* Bbuf = (_Float16*)(ws + 80 * MB);    // gate -> att_out (in-place)
  float*    R1   = (float*)   (ws + 112 * MB);
  float*    R2   = (float*)   (ws + 114 * MB);
  float*    SS   = (float*)   (ws + 116 * MB);
  float*    PP   = (float*)   (ws + 118 * MB);
  _Float16* ATTW = (_Float16*)(ws + 120 * MB);

  // ---- packed-weight arena at +124MB (~3.3MB) ----
  _Float16* WPK = (_Float16*)(ws + 124 * MB);
  size_t woff = 0;
  auto walloc = [&](int Cout, int Cin, int KK) -> _Float16* {
    _Float16* p = WPK + woff;
    woff += (size_t)((Cout + 15) / 16) * KK * (Cin / 32) * 512;
    return p;
  };
  auto packW = [&](const float* W, _Float16* P, int Cout, int Cin, int KK) {
    const int total = ((Cout + 15) / 16) * KK * (Cin / 32) * 32;
    pack_weights<<<(total + 255) / 256, 256, 0, stream>>>(W, P, Cout, Cin, KK);
  };
  _Float16* qwp = walloc(256, 256, 1);  packW(q_w, qwp, 256, 256, 1);
  _Float16* kwp = walloc(256, 256, 1);  packW(k_w, kwp, 256, 256, 1);
  _Float16* vwp = walloc(256, 256, 1);  packW(v_w, vwp, 256, 256, 1);
  _Float16* bowp = walloc(256, 256, 1); packW(b_out_w, bowp, 256, 256, 1);
  _Float16* owp = walloc(256, 256, 1);  packW(out_w, owp, 256, 256, 1);
  _Float16 *ruw1p[6], *ruw2p[6], *ruw3p[6];
  for (int j = 0; j < 6; ++j) {
    ruw1p[j] = walloc(128, 256, 1); packW(ru[j][0], ruw1p[j], 128, 256, 1);
    ruw2p[j] = walloc(128, 128, 9); packW(ru[j][2], ruw2p[j], 128, 128, 9);
    ruw3p[j] = walloc(256, 128, 1); packW(ru[j][4], ruw3p[j], 256, 128, 1);
  }
  _Float16* rw1p = walloc(8, 256, 9);   packW(r_w1, rw1p, 8, 256, 9);
  _Float16* rskp = walloc(8, 256, 1);   packW(r_skw, rskp, 8, 256, 1);

  const size_t ldsConv = (size_t)HW * 256 * sizeof(_Float16);  // 128KB
  const size_t ldsRU   = (size_t)98304 * sizeof(_Float16);     // 192KB (<=320KB/WGP)

  // q/k/v 1x1 convs (packed-input buffer reused serially on the stream)
  pack_nchw_to_phwc<<<32, 256, 0, stream>>>(query, PK, 256);
  conv_wmma<<<32, 256, ldsConv, stream>>>(PK, qwp, q_b, nullptr, QA, 256, 256, 0, 0, 0);
  pack_nchw_to_phwc<<<32, 256, 0, stream>>>(kin, PK, 256);
  conv_wmma<<<32, 256, ldsConv, stream>>>(PK, kwp, k_b, nullptr, KA, 256, 256, 0, 0, 0);
  pack_nchw_to_phwc<<<32, 256, 0, stream>>>(vin, PK, 256);
  conv_wmma<<<32, 256, ldsConv, stream>>>(PK, vwp, v_b, nullptr, VA, 256, 256, 0, 0, 0);

  // att_in = q[:, :, None] + k[:, None, :]  (256 images)
  pairsum<<<256, 256, 0, stream>>>(QA, KA, X0);

  // a-branch: 3 fused residual units
  residual_unit_wmma<<<256, 256, ldsRU, stream>>>(X0,   Abuf, ruw1p[0], ru[0][1], ruw2p[0], ru[0][3], ruw3p[0], ru[0][5]);
  residual_unit_wmma<<<256, 256, ldsRU, stream>>>(Abuf, Abuf, ruw1p[1], ru[1][1], ruw2p[1], ru[1][3], ruw3p[1], ru[1][5]);
  residual_unit_wmma<<<256, 256, ldsRU, stream>>>(Abuf, Abuf, ruw1p[2], ru[2][1], ruw2p[2], ru[2][3], ruw3p[2], ru[2][5]);
  // b-branch: 3 fused residual units + b_out 1x1 (no activation)
  residual_unit_wmma<<<256, 256, ldsRU, stream>>>(X0,   Bbuf, ruw1p[3], ru[3][1], ruw2p[3], ru[3][3], ruw3p[3], ru[3][5]);
  residual_unit_wmma<<<256, 256, ldsRU, stream>>>(Bbuf, Bbuf, ruw1p[4], ru[4][1], ruw2p[4], ru[4][3], ruw3p[4], ru[4][5]);
  residual_unit_wmma<<<256, 256, ldsRU, stream>>>(Bbuf, Bbuf, ruw1p[5], ru[5][1], ruw2p[5], ru[5][3], ruw3p[5], ru[5][5]);
  conv_wmma<<<256, 256, ldsConv, stream>>>(Bbuf, bowp, b_out_b, nullptr, Bbuf, 256, 256, 0, 0, 0);

  // att_out = x + a * sigmoid(b)   (in-place into Bbuf)
  combine_gate<<<256, 256, 0, stream>>>(X0, Abuf, Bbuf);

  // residual block: leaky(conv3x3 256->8) -> leaky(conv3x3 8->8), + 1x1 skip
  conv_wmma<<<256, 256, ldsConv, stream>>>(Bbuf, rw1p, r_b1, nullptr, R1, 256, 8, 1, 2, 1);
  conv3x3_c8_scalar<<<256, 256, 0, stream>>>(R1, r_w2, r_b2, R2);
  conv_wmma<<<256, 256, ldsConv, stream>>>(Bbuf, rskp, r_skb, nullptr, SS, 256, 8, 0, 0, 1);

  // softmax over KL, weighted sum with vv, final 1x1 conv -> f32 NCHW d_out
  score_softmax<<<32, 256, 0, stream>>>(R2, SS, PP);
  attn_apply<<<32, 256, 0, stream>>>(PP, VA, ATTW);
  conv_wmma<<<32, 256, ldsConv, stream>>>(ATTW, owp, out_b, nullptr, d_out, 256, 256, 0, 0, 2);
}
